// GATWeight_46849503265382
// MI455X (gfx1250) — matrix operand order repaired and checked by reading, not compiled, not run on hardware
//
#include <hip/hip_runtime.h>
#include <hip/hip_bf16.h>
#include <math.h>

// ---------------------------------------------------------------------------
// GAT forward for MI455X (gfx1250, wave32, WMMA + TDM).
// attn_tensor stream (805 MB @ 23.3 TB/s ~35us) is the roofline floor; agg
// (67 MB) stays L2-resident for the 4 later passes. All O(N^2 C) math uses
// v_wmma_f32_16x16x32_f16. Feature matrices are stored K-contiguous (H^T) so
// every WMMA B-fragment is two b128 loads; agg tiles are staged to LDS by the
// Tensor Data Mover (with D# padding giving a conflict-free 17-float stride).
// Scratch requirement ~72 MB.
// ---------------------------------------------------------------------------

typedef __attribute__((ext_vector_type(16))) _Float16 v16h;
typedef __attribute__((ext_vector_type(8)))  _Float16 v8h;
typedef __attribute__((ext_vector_type(8)))  float    v8f;
typedef __attribute__((ext_vector_type(4)))  unsigned int u32x4;
typedef __attribute__((ext_vector_type(8)))  int      i32x8;
typedef __attribute__((ext_vector_type(4)))  int      i32x4;

#define NN      4096
#define INCH    128
#define HIDC    64
#define NHEADS  2
#define OUTC    16
#define NGRAPH  8
#define NEGSL   0.2f

#if defined(__HIP_DEVICE_COMPILE__) && __has_builtin(__builtin_amdgcn_tensor_load_to_lds) && __has_builtin(__builtin_amdgcn_s_wait_tensorcnt)
#define USE_TDM 1
#else
#define USE_TDM 0
#endif

// ---------------------------------------------------------------------------
// Kernel 0: zero the small accumulators.
// ---------------------------------------------------------------------------
__global__ void init_zero_kernel(float* red, float* gsum, float* gcnt) {
    int t = threadIdx.x;
    if (t < 8) red[t] = 0.f;
    if (t < 8) gcnt[t] = 0.f;
    for (int i = t; i < NGRAPH * HIDC; i += blockDim.x) gsum[i] = 0.f;
}

// ---------------------------------------------------------------------------
// Kernel 1: agg[i][j] = sum_c attn[c][i][j]*w[c] + b  (float4 streaming),
// fused off-diagonal masked sum/count reduction for mean_e.
// ---------------------------------------------------------------------------
__global__ void agg_stream_kernel(const float* __restrict__ attn,
                                  const float* __restrict__ w12,
                                  const float* __restrict__ bptr,
                                  float* __restrict__ agg,
                                  float* __restrict__ red) {
    __shared__ float ws12[12];
    __shared__ float rs_[256];
    __shared__ float rc_[256];
    int tid = threadIdx.x;
    if (tid < 12) ws12[tid] = w12[tid];
    __syncthreads();

    long idx4   = (long)blockIdx.x * blockDim.x + tid;     // float4 units
    long plane4 = (long)NN * NN / 4;
    float b = bptr[0];
    float lsum = 0.f, lcnt = 0.f;

    const float4* base = reinterpret_cast<const float4*>(attn) + idx4;
    float4 acc; acc.x = b; acc.y = b; acc.z = b; acc.w = b;
#pragma unroll
    for (int c = 0; c < 12; ++c) {
        if (c + 1 < 12) __builtin_prefetch(base + (long)(c + 1) * plane4, 0, 0);
        float4 v = base[(long)c * plane4];
        float wc = ws12[c];
        acc.x += v.x * wc; acc.y += v.y * wc;
        acc.z += v.z * wc; acc.w += v.w * wc;
    }
    reinterpret_cast<float4*>(agg)[idx4] = acc;

    long e0 = idx4 * 4;
    int  i  = (int)(e0 / NN);
    int  j0 = (int)(e0 % NN);
    float vals[4] = {acc.x, acc.y, acc.z, acc.w};
#pragma unroll
    for (int e = 0; e < 4; ++e) {
        int j = j0 + e;
        if (i != j && vals[e] > 0.f) { lsum += vals[e]; lcnt += 1.f; }
    }
    rs_[tid] = lsum; rc_[tid] = lcnt;
    __syncthreads();
    for (int s = 128; s > 0; s >>= 1) {
        if (tid < s) { rs_[tid] += rs_[tid + s]; rc_[tid] += rc_[tid + s]; }
        __syncthreads();
    }
    if (tid == 0) { atomicAdd(&red[0], rs_[0]); atomicAdd(&red[1], rc_[0]); }
}

// ---------------------------------------------------------------------------
// Kernel 2: finalize mean_e and per-head edge constants k[h] = We.att_e.
// scal[0]=mean_e, scal[1..2]=k1[h], scal[3]=k2.
// ---------------------------------------------------------------------------
__global__ void finalize_kernel(const float* __restrict__ red,
                                float* __restrict__ scal,
                                const float* __restrict__ We1,
                                const float* __restrict__ ae1,
                                const float* __restrict__ We2,
                                const float* __restrict__ ae2) {
    if (threadIdx.x == 0 && blockIdx.x == 0) {
        scal[0] = red[0] / fmaxf(red[1], 1.f);
        float k0 = 0.f, k1 = 0.f, k2 = 0.f;
        for (int c = 0; c < HIDC; ++c) {
            k0 += We1[c] * ae1[c];
            k1 += We1[HIDC + c] * ae1[HIDC + c];
            k2 += We2[c] * ae2[c];
        }
        scal[1] = k0; scal[2] = k1; scal[3] = k2;
    }
}

// ---------------------------------------------------------------------------
// f32 -> f16 straight conversion (weights stay in their native [n][k] order:
// that IS the K-contiguous B layout WMMA wants).
// ---------------------------------------------------------------------------
__global__ void conv_f16_kernel(const float* __restrict__ a,
                                _Float16* __restrict__ o, int n) {
    int g = blockIdx.x * blockDim.x + threadIdx.x;
    if (g < n) o[g] = (_Float16)a[g];
}

// ---------------------------------------------------------------------------
// Kernel 3: WMMA GEMM  CT[n][m] = (A[M][K] * Bc[n][k]^T), one wave per tile.
// A frag : row = lane&15,  K = ko..ko+7, ko+16..ko+23  (two b128 loads)
// B frag : col = lane&15,  K = ko..ko+7, ko+16..ko+23  (two b128 loads, Bc K-contig)
// D frag : M = r + 8*(lane>>4), N = lane&15 -> transposed store is one b128.
// ---------------------------------------------------------------------------
__global__ void wmma_gemm_nt_kernel(const _Float16* __restrict__ A, int lda,
                                    const _Float16* __restrict__ Bc, int ldb,
                                    _Float16* __restrict__ CT, int ldct,
                                    int K) {
    int lane = threadIdx.x & 31;
    int m0 = blockIdx.x * 16, n0 = blockIdx.y * 16;
    int am = lane & 15, ak = (lane >> 4) * 8;
    int bn = lane & 15, bk = (lane >> 4) * 8;
    v8f acc = {};
    for (int k0 = 0; k0 < K; k0 += 32) {
        const _Float16* ap = A + (size_t)(m0 + am) * lda + k0 + ak;
        v8h alo = *reinterpret_cast<const v8h*>(ap);
        v8h ahi = *reinterpret_cast<const v8h*>(ap + 16);
        const _Float16* bp = Bc + (size_t)(n0 + bn) * ldb + k0 + bk;
        v8h blo = *reinterpret_cast<const v8h*>(bp);
        v8h bhi = *reinterpret_cast<const v8h*>(bp + 16);
        v16h a, b;
#pragma unroll
        for (int e = 0; e < 8; ++e) {
            a[e] = alo[e]; a[8 + e] = ahi[e];
            b[e] = blo[e]; b[8 + e] = bhi[e];
        }
        acc = __builtin_amdgcn_wmma_f32_16x16x32_f16(false, a, false, b,
                                                     (short)0, acc, false, false);
    }
    int cn = lane & 15, rb = (lane >> 4) * 8;
    v8h o;
#pragma unroll
    for (int r = 0; r < 8; ++r) o[r] = (_Float16)acc[r];
    *reinterpret_cast<v8h*>(CT + (size_t)(n0 + cn) * ldct + m0 + rb) = o;
}

// ---------------------------------------------------------------------------
// Kernel 4: per-node attention dots from transposed features HT[C][NN].
// Lane-adjacent i => fully coalesced per channel row.
// ---------------------------------------------------------------------------
__global__ void head_dots_t_kernel(const _Float16* __restrict__ HT,
                                   const float* __restrict__ att_src,
                                   const float* __restrict__ att_dst,
                                   float* __restrict__ ssrc,
                                   float* __restrict__ sdst,
                                   int heads) {
    int i = blockIdx.x * blockDim.x + threadIdx.x;
    if (i >= NN) return;
    for (int h = 0; h < heads; ++h) {
        float a = 0.f, d = 0.f;
#pragma unroll 8
        for (int c = 0; c < HIDC; ++c) {
            float v = (float)HT[(size_t)(h * HIDC + c) * NN + i];
            a += v * att_src[h * HIDC + c];
            d += v * att_dst[h * HIDC + c];
        }
        ssrc[h * NN + i] = a;
        sdst[h * NN + i] = d;
    }
}

// ---------------------------------------------------------------------------
// Kernel 5: online-softmax column stats (max M[j], denom S[j]) per head.
// ---------------------------------------------------------------------------
__device__ inline void osm_merge(float& M, float& S, float m2, float s2) {
    if (m2 > M) { S = S * __expf(M - m2) + s2; M = m2; }
    else        { S += s2 * __expf(m2 - M); }
}

__global__ void gat_stats_kernel(const float* __restrict__ agg,
                                 const float* __restrict__ scal,
                                 const float* __restrict__ ssrc,
                                 const float* __restrict__ sdst,
                                 const float* __restrict__ kvec,
                                 float* __restrict__ Mst,
                                 float* __restrict__ Sden) {
    __shared__ float mm[256], ss[256];
    int h   = blockIdx.y;
    int tid = threadIdx.x;
    int jl  = tid & 31, io = tid >> 5;
    int j   = blockIdx.x * 32 + jl;
    float sdj = sdst[h * NN + j];
    float kh  = kvec[h];
    float mE  = scal[0];
    float M = -1e30f, S = 0.f;
    for (int i = io; i < NN; i += 8) {
        float ev = agg[(size_t)i * NN + j];
        bool  dg = (i == j);
        float e  = dg ? mE : ev;
        bool  ok = dg || (ev > 0.f);
        if (!ok) continue;
        float sc = ssrc[h * NN + i] + sdj + e * kh;
        sc = sc > 0.f ? sc : NEGSL * sc;
        if (sc > M) { S = S * __expf(M - sc) + 1.f; M = sc; }
        else        { S += __expf(sc - M); }
    }
    mm[tid] = M; ss[tid] = S;
    __syncthreads();
    if (tid < 32) {
        float Mo = mm[tid], So = ss[tid];
#pragma unroll
        for (int r = 1; r < 8; ++r) osm_merge(Mo, So, mm[tid + 32 * r], ss[tid + 32 * r]);
        Mst[h * NN + j]  = Mo;
        Sden[h * NN + j] = So;
    }
}

// ---------------------------------------------------------------------------
// Kernel 6: fused GAT aggregation GEMM:
//   out[j][c] = bias[c] + (1/S[j]) * sum_i exp(lrelu(score(i,j)) - M[j]) * H[i][c]
// Alpha-numerator tiles built in-register from a TDM-staged agg tile in LDS
// (D# pad gives the conflict-free 17-float row stride). Features come straight
// from global HT (K-contiguous) as b128 fragment loads.
// Block = heads*4 waves; one 16(j) x 16(c) WMMA tile per wave.
// ---------------------------------------------------------------------------
__global__ void gat_aggregate_kernel(const float* __restrict__ agg,
                                     const float* __restrict__ scal,
                                     const float* __restrict__ ssrc,
                                     const float* __restrict__ sdst,
                                     const float* __restrict__ kvec,
                                     const float* __restrict__ Mst,
                                     const float* __restrict__ Sden,
                                     const _Float16* __restrict__ HT,
                                     const float* __restrict__ bias,
                                     _Float16* __restrict__ outF16,
                                     float* __restrict__ outF32, int ldo,
                                     int heads, int relu_flag) {
    __shared__ __align__(16) float evt[32][17];        // agg tile [i][j], stride 17
    __shared__ float ssc[NHEADS * 32];                 // s_src chunk

    int nth  = blockDim.x;
    int tid  = threadIdx.x;
    int lane = tid & 31;
    int wave = tid >> 5;
    int c0   = wave * 16;          // output feature tile
    int head = c0 >> 6;            // HIDC == 64
    int j0   = blockIdx.x * 16;

    int am = lane & 15, ak = (lane >> 4) * 8;
    int bn = lane & 15, bk = (lane >> 4) * 8;
    int jA = j0 + am;

    float sdj = sdst[head * NN + jA];
    float Mj  = Mst[head * NN + jA];
    float kh  = kvec[head];
    float mE  = scal[0];

#if USE_TDM
    // Invariant D# group 1: 4-byte elements, pad 1 dword after every 16 dwords
    // (LDS row stride 17 floats), tile 16(x) x 32(y), tensor 4096x4096,
    // dim0 stride 4096 elements.
    i32x8 g1;
    g1[0] = (2 << 16) | (1 << 20) | (3 << 22);   // data_size=4B, pad_en, interval=16dw (amount=1dw)
    g1[1] = (int)(4096u << 16);                  // tensor_dim0 = 4096 (low16 in [31:16])
    g1[2] = (int)(4096u << 16);                  // tensor_dim1 = 4096 (low16 in [31:16])
    g1[3] = (int)(16u << 16);                    // tile_dim0 = 16
    g1[4] = 32;                                  // tile_dim1 = 32, tile_dim2 = 0
    g1[5] = 4096;                                // tensor_dim0_stride = 4096 (elements)
    g1[6] = 0;
    g1[7] = 0;
    i32x4 gz4 = {0, 0, 0, 0};
    i32x8 gz8 = {0, 0, 0, 0, 0, 0, 0, 0};
    unsigned lds_evt = (unsigned)(uintptr_t)(&evt[0][0]);  // flat->LDS: low 32 bits
#endif

    v8f acc = {};

    for (int i0 = 0; i0 < NN; i0 += 32) {
#if USE_TDM
        if (wave == 0) {
            unsigned long long ga =
                (unsigned long long)(uintptr_t)(agg + (size_t)i0 * NN + j0);
            u32x4 g0;
            g0[0] = 1u;                                        // count=1
            g0[1] = lds_evt;                                   // lds_addr
            g0[2] = (unsigned)(ga & 0xFFFFFFFFu);              // global_addr[31:0]
            g0[3] = (unsigned)((ga >> 32) & 0x01FFFFFFu) | (2u << 30);  // addr[56:32]|type=2
            __builtin_amdgcn_tensor_load_to_lds(g0, g1, gz4, gz4, gz8, 0);
        }
#else
        for (int idx = tid; idx < 512; idx += nth) {
            int r = idx >> 4, cj = idx & 15;
            evt[r][cj] = agg[(size_t)(i0 + r) * NN + j0 + cj];
        }
#endif
        for (int idx = tid; idx < heads * 32; idx += nth) {
            int hh = idx >> 5, rr = idx & 31;
            ssc[hh * 32 + rr] = ssrc[hh * NN + i0 + rr];
        }
#if USE_TDM
        if (wave == 0) __builtin_amdgcn_s_wait_tensorcnt((short)0);
#endif
        __syncthreads();

        v16h a, b;
#pragma unroll
        for (int e = 0; e < 16; ++e) {
            int kk   = ak + e + (e >= 8 ? 8 : 0);      // K index within 32
            int iabs = i0 + kk;
            float ev = evt[kk][am];
            bool  dg = (iabs == jA);
            float eV = dg ? mE : ev;
            bool  ok = dg || (ev > 0.f);
            float sc = ssc[head * 32 + kk] + sdj + eV * kh;
            sc = sc > 0.f ? sc : NEGSL * sc;
            float p  = ok ? __expf(sc - Mj) : 0.f;
            a[e] = (_Float16)p;
        }
        const _Float16* bp = HT + (size_t)(c0 + bn) * NN + i0 + bk;
        v8h blo = *reinterpret_cast<const v8h*>(bp);
        v8h bhi = *reinterpret_cast<const v8h*>(bp + 16);
#pragma unroll
        for (int e = 0; e < 8; ++e) { b[e] = blo[e]; b[8 + e] = bhi[e]; }

        acc = __builtin_amdgcn_wmma_f32_16x16x32_f16(false, a, false, b,
                                                     (short)0, acc, false, false);
        __syncthreads();
    }

    int cn = lane & 15, rb = (lane >> 4) * 8;
    float bz = bias[c0 + cn];
#pragma unroll
    for (int r = 0; r < 8; ++r) {
        int jabs = j0 + r + rb;
        float den = Sden[head * NN + jabs];
        float v = acc[r] / den + bz;
        if (relu_flag) v = fmaxf(v, 0.f);
        if (outF16) outF16[(size_t)jabs * ldo + c0 + cn] = (_Float16)v;
        if (outF32) outF32[(size_t)jabs * ldo + c0 + cn] = v;
    }
}

// ---------------------------------------------------------------------------
// Kernel 7: per-graph mean pooling (atomic accumulate).
// ---------------------------------------------------------------------------
__global__ void pool_kernel(const float* __restrict__ h,
                            const int* __restrict__ bidx,
                            float* __restrict__ gsum,
                            float* __restrict__ gcnt) {
    int i = blockIdx.x * blockDim.x + threadIdx.x;
    if (i >= NN) return;
    int b = bidx[i];
    atomicAdd(&gcnt[b], 1.f);
    for (int c = 0; c < HIDC; ++c)
        atomicAdd(&gsum[b * HIDC + c], h[(size_t)i * HIDC + c]);
}

// ---------------------------------------------------------------------------
// Kernel 8: MLP head: Linear(64,64)+ReLU -> LayerNorm -> Linear(64,16).
// ---------------------------------------------------------------------------
__global__ void head_kernel(const float* __restrict__ gsum,
                            const float* __restrict__ gcnt,
                            const float* __restrict__ l1w,
                            const float* __restrict__ l1b,
                            const float* __restrict__ lng,
                            const float* __restrict__ lnb,
                            const float* __restrict__ l2w,
                            const float* __restrict__ l2b,
                            float* __restrict__ out) {
    __shared__ float g[NGRAPH][HIDC];
    __shared__ float y[NGRAPH][HIDC];
    __shared__ float mu[NGRAPH], rs[NGRAPH];
    int t = threadIdx.x;
    int bq = t >> 6, c = t & 63;
    float cnt = fmaxf(gcnt[bq], 1.f);
    g[bq][c] = gsum[bq * HIDC + c] / cnt;
    __syncthreads();
    float acc = l1b[c];
    for (int k = 0; k < HIDC; ++k) acc += g[bq][k] * l1w[c * HIDC + k];
    acc = fmaxf(acc, 0.f);
    y[bq][c] = acc;
    __syncthreads();
    if (c == 0) {
        float m = 0.f;
        for (int k = 0; k < HIDC; ++k) m += y[bq][k];
        m /= HIDC;
        float v = 0.f;
        for (int k = 0; k < HIDC; ++k) { float d = y[bq][k] - m; v += d * d; }
        v /= HIDC;
        mu[bq] = m; rs[bq] = rsqrtf(v + 1e-5f);
    }
    __syncthreads();
    y[bq][c] = (y[bq][c] - mu[bq]) * rs[bq] * lng[c] + lnb[c];
    __syncthreads();
    if (t < NGRAPH * OUTC) {
        int b2 = t >> 4, o = t & 15;
        float a = l2b[o];
        for (int k = 0; k < HIDC; ++k) a += y[b2][k] * l2w[o * HIDC + k];
        out[b2 * OUTC + o] = a;
    }
}

// ---------------------------------------------------------------------------
// Host launcher.
// ---------------------------------------------------------------------------
extern "C" void kernel_launch(void* const* d_in, const int* in_sizes, int n_in,
                              void* d_out, int out_size, void* d_ws, size_t ws_size,
                              hipStream_t stream) {
    (void)in_sizes; (void)n_in; (void)out_size; (void)ws_size;
    const float* x    = (const float*)d_in[0];
    const float* attn = (const float*)d_in[1];
    const int*   bidx = (const int*)d_in[2];
    const float* aggw = (const float*)d_in[3];
    const float* aggb = (const float*)d_in[4];
    const float* W1   = (const float*)d_in[5];
    const float* as1  = (const float*)d_in[6];
    const float* ad1  = (const float*)d_in[7];
    const float* We1  = (const float*)d_in[8];
    const float* ae1  = (const float*)d_in[9];
    const float* b1   = (const float*)d_in[10];
    const float* W2   = (const float*)d_in[11];
    const float* as2  = (const float*)d_in[12];
    const float* ad2  = (const float*)d_in[13];
    const float* We2  = (const float*)d_in[14];
    const float* ae2  = (const float*)d_in[15];
    const float* b2   = (const float*)d_in[16];
    const float* l1w  = (const float*)d_in[17];
    const float* l1b  = (const float*)d_in[18];
    const float* lng  = (const float*)d_in[19];
    const float* lnb  = (const float*)d_in[20];
    const float* l2w  = (const float*)d_in[21];
    const float* l2b  = (const float*)d_in[22];
    float* out = (float*)d_out;

    char* ws = (char*)d_ws;
    size_t off = 0;
    auto take = [&](size_t bytes) -> char* {
        char* p = ws + off;
        off = (off + bytes + 255) & ~(size_t)255;
        return p;
    };
    float*    agg  = (float*)take((size_t)NN * NN * 4);        // 67.1 MB
    float*    red  = (float*)take(8 * 4);
    float*    scal = (float*)take(8 * 4);
    _Float16* Xh   = (_Float16*)take((size_t)NN * INCH * 2);
    _Float16* W1h  = (_Float16*)take((size_t)INCH * INCH * 2); // [n][k] K-contig
    _Float16* W2h  = (_Float16*)take((size_t)HIDC * INCH * 2); // [n][k] K-contig
    _Float16* H1T  = (_Float16*)take((size_t)INCH * NN * 2);   // [c][i]
    float*    ss1  = (float*)take((size_t)NHEADS * NN * 4);
    float*    sd1  = (float*)take((size_t)NHEADS * NN * 4);
    float*    M1   = (float*)take((size_t)NHEADS * NN * 4);
    float*    S1   = (float*)take((size_t)NHEADS * NN * 4);
    _Float16* O1   = (_Float16*)take((size_t)NN * INCH * 2);   // [j][c] row-major
    _Float16* H2T  = (_Float16*)take((size_t)HIDC * NN * 2);   // [c][i]
    float*    ss2  = (float*)take((size_t)NN * 4);
    float*    sd2  = (float*)take((size_t)NN * 4);
    float*    M2   = (float*)take((size_t)NN * 4);
    float*    S2   = (float*)take((size_t)NN * 4);
    float*    O2   = (float*)take((size_t)NN * HIDC * 4);      // [j][c] f32
    float*    gsum = (float*)take((size_t)NGRAPH * HIDC * 4);
    float*    gcnt = (float*)take((size_t)NGRAPH * 4);

    // 0. zero accumulators
    init_zero_kernel<<<1, 512, 0, stream>>>(red, gsum, gcnt);

    // 1. aggregated adjacency + mean-edge reduction (streams 805 MB once)
    long total4 = (long)NN * NN / 4;
    agg_stream_kernel<<<(unsigned)(total4 / 256), 256, 0, stream>>>(attn, aggw, aggb, agg, red);

    // 2. mean_e and edge constants
    finalize_kernel<<<1, 32, 0, stream>>>(red, scal, We1, ae1, We2, ae2);

    // 3. f16 conversions (weights keep native [n][k] = K-contiguous B layout)
    conv_f16_kernel<<<(NN * INCH + 255) / 256, 256, 0, stream>>>(x, Xh, NN * INCH);
    conv_f16_kernel<<<(INCH * INCH + 255) / 256, 256, 0, stream>>>(W1, W1h, INCH * INCH);
    conv_f16_kernel<<<(HIDC * INCH + 255) / 256, 256, 0, stream>>>(W2, W2h, HIDC * INCH);

    // 4. H1T = (x @ W1^T)^T   (4096x128x128, WMMA, transposed b128 store)
    wmma_gemm_nt_kernel<<<dim3(NN / 16, INCH / 16), 32, 0, stream>>>(
        Xh, INCH, W1h, INCH, H1T, NN, INCH);

    // 5. attention dots layer 1
    head_dots_t_kernel<<<NN / 256, 256, 0, stream>>>(H1T, as1, ad1, ss1, sd1, NHEADS);

    // 6. softmax stats layer 1
    gat_stats_kernel<<<dim3(NN / 32, NHEADS), 256, 0, stream>>>(
        agg, scal, ss1, sd1, scal + 1, M1, S1);

    // 7. aggregation layer 1 (+bias +ReLU) -> O1 (f16 row-major)
    gat_aggregate_kernel<<<NN / 16, NHEADS * 128, 0, stream>>>(
        agg, scal, ss1, sd1, scal + 1, M1, S1, H1T, b1, O1, nullptr, INCH,
        NHEADS, 1);

    // 8. H2T = (relu(O1) @ W2^T)^T  (4096x64x128, WMMA)
    wmma_gemm_nt_kernel<<<dim3(NN / 16, HIDC / 16), 32, 0, stream>>>(
        O1, INCH, W2h, INCH, H2T, NN, INCH);

    // 9. attention dots layer 2
    head_dots_t_kernel<<<NN / 256, 256, 0, stream>>>(H2T, as2, ad2, ss2, sd2, 1);

    // 10. softmax stats layer 2
    gat_stats_kernel<<<dim3(NN / 32, 1), 256, 0, stream>>>(
        agg, scal, ss2, sd2, scal + 3, M2, S2);

    // 11. aggregation layer 2 (+bias) -> O2 (f32 row-major)
    gat_aggregate_kernel<<<NN / 16, 128, 0, stream>>>(
        agg, scal, ss2, sd2, scal + 3, M2, S2, H2T, b2, nullptr, O2, HIDC,
        1, 0);

    // 12. per-graph mean pool
    pool_kernel<<<NN / 256, 256, 0, stream>>>(O2, bidx, gsum, gcnt);

    // 13. MLP head -> d_out [8,16]
    head_kernel<<<1, 512, 0, stream>>>(gsum, gcnt, l1w, l1b, lng, lnb, l2w, l2b, out);
}
